// SubMDTA_36850819400310
// MI455X (gfx1250) — matrix operand order repaired and checked by reading, not compiled, not run on hardware
//
#include <hip/hip_runtime.h>
#include <hip/hip_bf16.h>

// ---------------------------------------------------------------------------
// GIN forward on MI455X (gfx1250, wave32).
// GEMMs run on v_wmma_f32_16x16x32_bf16 (bf16 in, f32 accumulate).
// Weights are staged in LDS pre-swizzled into B-fragment order so each
// fragment is two ds_load_b128 into contiguous VGPR quads; B fragments are
// double-buffered so each WMMA waits only on loads issued one step earlier.
// Scatter/BN/pool stay f32 (memory-bound, L2-resident working set).
// ---------------------------------------------------------------------------

#define NN 50000
#define NE 600000
#define NG 512
#define DIM 128
#define NF 78
#define NL 4
#define BN_EPS 1e-5f

typedef __bf16 v16bf __attribute__((ext_vector_type(16)));
typedef float  v8f   __attribute__((ext_vector_type(8)));

typedef union { v16bf v; unsigned int u[8]; uint4 q[2]; } Frag16;

// f32 -> bf16 bits, round-to-nearest-even (inputs are finite).
__device__ __forceinline__ unsigned short f2bf(float f) {
  unsigned int u = __builtin_bit_cast(unsigned int, f);
  unsigned int r = 0x7FFFu + ((u >> 16) & 1u);
  return (unsigned short)((u + r) >> 16);
}

// ---------------------------------------------------------------------------
// WMMA GEMM:  out[N x 128] = act( A[N x KT*32](bf16) @ W[Kreal x 128](f32) + b )
// 256 threads = 8 waves; wave w computes rows [blk*128 + w*16, +16), all 128 cols.
//
// LDS B layout (fragment order): for chunk kk, column tile c, lane l, the 8
// packed bf16-pair dwords (VGPR j -> K = 2j + 16*(l>>4), N = c*16 + (l&15))
// live at Blds[((kk*8+c)*32 + l)*12 + j].  Lane stride 12 dwords (48 B) keeps
// 16B alignment and makes each 16-lane b128 phase bank-conflict-free.
// ---------------------------------------------------------------------------
template <int KT, bool RELU, bool OUT_BF16>
__global__ __launch_bounds__(256)
void gemm_wmma(const unsigned short* __restrict__ A,
               const float* __restrict__ W,
               const float* __restrict__ bias,
               float* __restrict__ outF,
               unsigned short* __restrict__ outB,
               int N, int Kreal) {
  __shared__ __align__(16) unsigned int Blds[KT * 8 * 32 * 12];

  const int tid = threadIdx.x;
  // Stage weights into fragment-ordered LDS (bf16 pairs packed in dwords).
  for (int i = tid; i < KT * 2048; i += 256) {
    int j    = i & 7;
    int lane = (i >> 3) & 31;
    int c    = (i >> 8) & 7;
    int kk   = i >> 11;
    int k    = kk * 32 + (lane >> 4) * 16 + j * 2;
    int n    = c * 16 + (lane & 15);
    float w0 = (k     < Kreal) ? W[(size_t)k * 128 + n]       : 0.f;
    float w1 = (k + 1 < Kreal) ? W[(size_t)(k + 1) * 128 + n] : 0.f;
    Blds[(((kk * 8 + c) * 32 + lane) * 12) + j] =
        (unsigned int)f2bf(w0) | ((unsigned int)f2bf(w1) << 16);
  }
  __syncthreads();

  const int wave = tid >> 5;
  const int lane = tid & 31;
  const int half = lane >> 4;     // lane half selects K sub-range (ISA 7.12.2)
  const int lm   = lane & 15;
  const int r0   = blockIdx.x * 128 + wave * 16;

  int arow = r0 + lm;             // A-fragment row for this lane (M = lane&15)
  if (arow > N - 1) arow = N - 1; // clamp; stores are guarded
  const unsigned int* Arow = (const unsigned int*)(A + (size_t)arow * (KT * 32));
  const unsigned int* Bl   = Blds + lane * 12;

  v8f acc[8];
  v8f zacc = {};
#pragma unroll
  for (int c = 0; c < 8; ++c) acc[c] = zacc;

#pragma unroll
  for (int kk = 0; kk < KT; ++kk) {
    // A 16x32 bf16 fragment: VGPR j holds K = 2*j + 16*half.
    // j=0..3 and j=4..7 are each 4 consecutive dwords -> two b128 loads.
    Frag16 a;
    a.q[0] = *(const uint4*)(Arow + kk * 16 + half * 8);
    a.q[1] = *(const uint4*)(Arow + kk * 16 + half * 8 + 4);

    // Double-buffered B fragments: issue loads for c+1 before the WMMA on c,
    // so each WMMA only needs dscnt<=2 (in-order DS returns).
    Frag16 bf[2];
    {
      const unsigned int* bp = Bl + (kk * 8) * 384;       // 32 lanes * 12 dwords
      bf[0].q[0] = *(const uint4*)(bp);
      bf[0].q[1] = *(const uint4*)(bp + 4);
    }
#pragma unroll
    for (int c = 0; c < 8; ++c) {
      if (c + 1 < 8) {
        const unsigned int* bp = Bl + (kk * 8 + c + 1) * 384;
        bf[(c + 1) & 1].q[0] = *(const uint4*)(bp);
        bf[(c + 1) & 1].q[1] = *(const uint4*)(bp + 4);
      }
      acc[c] = __builtin_amdgcn_wmma_f32_16x16x32_bf16(
          false, a.v, false, bf[c & 1].v, (short)0, acc[c], false, false);
    }
  }

  // D layout: VGPR r -> M = r + 8*half, N = (lane&15) + c*16
#pragma unroll
  for (int c = 0; c < 8; ++c) {
    int ncol = c * 16 + lm;
    float bv = bias[ncol];
#pragma unroll
    for (int r = 0; r < 8; ++r) {
      int row = r0 + r + half * 8;
      if (row < N) {
        float v = acc[c][r] + bv;
        if (RELU) v = fmaxf(v, 0.f);
        if (OUT_BF16) outB[(size_t)row * 128 + ncol] = f2bf(v);
        else          outF[(size_t)row * 128 + ncol] = v;
      }
    }
  }
}

// ---------------------------------------------------------------------------
// Elementwise / reduction helpers
// ---------------------------------------------------------------------------
__global__ void k_zero(float* __restrict__ p, int n) {
  int i = blockIdx.x * 256 + threadIdx.x;
  if (i < n) p[i] = 0.f;
}

__global__ void k_cvt_pad_x(const float* __restrict__ x,
                            unsigned short* __restrict__ xb, int n) {
  int i = blockIdx.x * 256 + threadIdx.x;
  if (i < n) {
    int node = i / 96, k = i - node * 96;
    float v = (k < NF) ? x[(size_t)node * NF + k] : 0.f;
    xb[i] = f2bf(v);
  }
}

__global__ void k_copy4(const float4* __restrict__ a, float4* __restrict__ b, int n4) {
  int i = blockIdx.x * 256 + threadIdx.x;
  if (i < n4) b[i] = a[i];
}

__global__ void k_cvt_bf16(const float* __restrict__ a,
                           unsigned short* __restrict__ b, int n) {
  int i = blockIdx.x * 256 + threadIdx.x;
  if (i < n) b[i] = f2bf(a[i]);
}

// z[dst] += h[src] ; one lane per (edge, feature), 2 edges per 256-thread block
__global__ void k_scatter(const float* __restrict__ h, const int* __restrict__ src,
                          const int* __restrict__ dst, float* __restrict__ z) {
  int i = blockIdx.x * 256 + threadIdx.x;
  int e = i >> 7, f = i & 127;
  if (e < NE) {
    int s = src[e], d = dst[e];
    atomicAdd(z + (size_t)d * 128 + f, h[(size_t)s * 128 + f]);
  }
}

// per-column sum / sum-of-squares partials -> atomics into stats[0:128]/[128:256]
__global__ void k_stats(const float* __restrict__ u, float* __restrict__ stats) {
  int f = threadIdx.x;                 // 128 threads
  int r0 = blockIdx.x * 128;
  float s = 0.f, s2 = 0.f;
  for (int r = 0; r < 128; ++r) {
    int row = r0 + r;
    if (row < NN) {
      float v = u[(size_t)row * 128 + f];
      s += v; s2 += v * v;
    }
  }
  atomicAdd(&stats[f], s);
  atomicAdd(&stats[128 + f], s2);
}

// BatchNorm (training-mode batch stats, biased var) + layer-weighted accumulate
__global__ void k_bn(const float* __restrict__ u, const float* __restrict__ stats,
                     const float* __restrict__ gamma, const float* __restrict__ beta,
                     const float* __restrict__ layer_w, int layer,
                     float* __restrict__ h, float* __restrict__ acc, int n) {
  int i = blockIdx.x * 256 + threadIdx.x;
  if (i < n) {
    int f = i & 127;
    const float invN = 1.f / (float)NN;
    float mean = stats[f] * invN;
    float var  = stats[128 + f] * invN - mean * mean;
    float rs   = rsqrtf(var + BN_EPS);
    float hv   = (u[i] - mean) * rs * gamma[f] + beta[f];
    h[i] = hv;
    acc[i] += layer_w[layer] * hv;
  }
}

// global_add_pool over graphs (+ conv bias broadcast)
__global__ void k_pool(const float* __restrict__ acc, const int* __restrict__ batch,
                       const float* __restrict__ layer_b, float* __restrict__ out, int n) {
  int i = blockIdx.x * 256 + threadIdx.x;
  if (i < n) {
    int node = i >> 7, f = i & 127;
    atomicAdd(&out[(size_t)batch[node] * 128 + f], acc[i] + layer_b[0]);
  }
}

// ---------------------------------------------------------------------------
// Orchestration
// ---------------------------------------------------------------------------
extern "C" void kernel_launch(void* const* d_in, const int* in_sizes, int n_in,
                              void* d_out, int out_size, void* d_ws, size_t ws_size,
                              hipStream_t stream) {
  const float* x       = (const float*)d_in[0];
  const int*   ei      = (const int*)d_in[1];
  const int*   batch   = (const int*)d_in[2];
  const float* ini_w1  = (const float*)d_in[3];
  const float* ini_b1  = (const float*)d_in[4];
  const float* ini_w2  = (const float*)d_in[5];
  const float* ini_b2  = (const float*)d_in[6];
  const float* gin_w1  = (const float*)d_in[7];
  const float* gin_b1  = (const float*)d_in[8];
  const float* gin_w2  = (const float*)d_in[9];
  const float* gin_b2  = (const float*)d_in[10];
  const float* gamma   = (const float*)d_in[11];
  const float* beta    = (const float*)d_in[12];
  const float* layer_w = (const float*)d_in[13];
  const float* layer_b = (const float*)d_in[14];
  float* out = (float*)d_out;

  const int* src = ei;
  const int* dst = ei + NE;

  // workspace layout (all sizes multiples of 256B)
  char* w = (char*)d_ws;
  unsigned short* xb = (unsigned short*)(w + 0);          //  9.6 MB  bf16 padded x [NN x 96]
  unsigned short* t  = (unsigned short*)(w + 9600000);    // 12.8 MB  bf16 mlp-mid  [NN x 128]
  unsigned short* zb = (unsigned short*)(w + 22400000);   // 12.8 MB  bf16 z        [NN x 128]
  float* h    = (float*)(w + 35200000);                   // 25.6 MB  node state
  float* z    = (float*)(w + 60800000);                   // 25.6 MB  z / u (reused)
  float* acc  = (float*)(w + 86400000);                   // 25.6 MB  layer-weighted sum
  float* stat = (float*)(w + 112000000);                  //  1 KB    BN sums

  const int NEL  = NN * 128;                 // node elements
  const int GB_E = (NEL + 255) / 256;        // elementwise grid
  const int GB_4 = (NEL / 4 + 255) / 256;    // float4 copy grid
  const int GB_G = (NN + 127) / 128;         // gemm / stats row-block grid
  const int GB_S = (NE * 128) / 256;         // scatter grid

  // init accumulators / output (d_out & d_ws are poisoned by harness)
  k_zero<<<GB_E, 256, 0, stream>>>(acc, NEL);
  k_zero<<<(NG * 128 + 255) / 256, 256, 0, stream>>>(out, NG * 128);

  // ini_embed: h = relu(x @ w1 + b1) @ w2 + b2
  k_cvt_pad_x<<<(NN * 96 + 255) / 256, 256, 0, stream>>>(x, xb, NN * 96);
  gemm_wmma<3, true,  true ><<<GB_G, 256, 0, stream>>>(xb, ini_w1, ini_b1, nullptr, t, NN, NF);
  gemm_wmma<4, false, false><<<GB_G, 256, 0, stream>>>(t, ini_w2, ini_b2, h, nullptr, NN, 128);

  for (int i = 0; i < NL; ++i) {
    // z = h + segment_sum(h[src] -> dst)
    k_copy4<<<GB_4, 256, 0, stream>>>((const float4*)h, (float4*)z, NEL / 4);
    k_scatter<<<GB_S, 256, 0, stream>>>(h, src, dst, z);
    k_cvt_bf16<<<GB_E, 256, 0, stream>>>(z, zb, NEL);

    // GIN MLP: u = relu( relu(z @ w1 + b1) @ w2 + b2 )   (u reuses z buffer)
    gemm_wmma<4, true, true ><<<GB_G, 256, 0, stream>>>(
        zb, gin_w1 + (size_t)i * 128 * 128, gin_b1 + i * 128, nullptr, t, NN, 128);
    gemm_wmma<4, true, false><<<GB_G, 256, 0, stream>>>(
        t,  gin_w2 + (size_t)i * 128 * 128, gin_b2 + i * 128, z, nullptr, NN, 128);

    // BatchNorm (batch stats) + acc += layer_w[i] * h
    k_zero<<<1, 256, 0, stream>>>(stat, 256);
    k_stats<<<GB_G, 128, 0, stream>>>(z, stat);
    k_bn<<<GB_E, 256, 0, stream>>>(z, stat, gamma + i * 128, beta + i * 128,
                                   layer_w, i, h, acc, NEL);
  }

  // pos_global_graph = segment_sum(acc + layer_b, batch)
  k_pool<<<GB_E, 256, 0, stream>>>(acc, batch, layer_b, out, NEL);
}
